// DistributedPiKVMoE_74217034875089
// MI455X (gfx1250) — matrix operand order repaired and checked by reading, not compiled, not run on hardware
//
#include <hip/hip_runtime.h>
#include <hip/hip_bf16.h>
#include <cstdint>

// ---------------------------------------------------------------------------
// DistributedPiKVMoE forward for MI455X (gfx1250, wave32, WMMA).
//
// Algebraic eliminations (exact):
//  * importance.mean() == 1.0  (softmax rows sum to 1)  -> attention path dead.
//  * new_values row0 and cached-value means fold to H-vectors by linearity.
// Hot loop: expert_output = sum_e routing[:,e] * (x @ We_e) (+bias/LoRA/mean),
// 137 GFLOP, compute-bound -> bf16 WMMA v_wmma_f32_16x16x32_bf16, fp32 accum.
// Software-pipelined, double-buffered LDS B staging via CDNA5
// global_load_async_to_lds_b128 (ASYNCcnt) with reg-staged fallback; A
// fragment pipelined one K-block ahead; one barrier per K-block.
// ---------------------------------------------------------------------------

#define Mm 16384      // B*S tokens
#define Hh 1024
#define Ee 4
#define Cc 1024
#define Rr 4
#define KEEPc 512

#define AS1 __attribute__((address_space(1)))
#define AS3 __attribute__((address_space(3)))

#if __has_builtin(__builtin_amdgcn_global_load_async_to_lds_b128) && \
    __has_builtin(__builtin_amdgcn_s_wait_asynccnt)
#define USE_ASYNC_LDS 1
#else
#define USE_ASYNC_LDS 0
#endif

typedef __attribute__((ext_vector_type(16))) __bf16 v16bf;
typedef __attribute__((ext_vector_type(8)))  __bf16 v8bf;
typedef __attribute__((ext_vector_type(8)))  float  v8f;
typedef int v4i32 __attribute__((vector_size(16)));  // matches builtin's V4i

union V16U { v16bf v; v8bf h[2]; };

__device__ __forceinline__ v16bf ld16x2(const __bf16* lo, const __bf16* hi) {
  V16U u;
  u.h[0] = *(const v8bf*)lo;
  u.h[1] = *(const v8bf*)hi;
  return u.v;
}

__device__ __forceinline__ v8f vzero8() {
  v8f z = {0.f, 0.f, 0.f, 0.f, 0.f, 0.f, 0.f, 0.f};
  return z;
}

#if USE_ASYNC_LDS
// ISA §10.2: flat LDS addresses map to LDS offset = addr[31:0]; global flat
// addresses are identity-mapped -> integer round-trip casts are exact.
__device__ __forceinline__ AS3 v4i32* to_lds(void* p) {
  return (AS3 v4i32*)(uint32_t)(uintptr_t)p;
}
__device__ __forceinline__ AS1 v4i32* to_glob(const void* p) {
  return (AS1 v4i32*)(uintptr_t)p;
}
#endif

// --------------------------- x -> bf16 -------------------------------------
__global__ __launch_bounds__(256) void k_cvt_bf16(const float* __restrict__ src,
                                                  __bf16* __restrict__ dst) {
  const int i = (blockIdx.x * 256 + threadIdx.x) * 4;
  const float4 f = *(const float4*)(src + i);
  dst[i + 0] = (__bf16)f.x;
  dst[i + 1] = (__bf16)f.y;
  dst[i + 2] = (__bf16)f.z;
  dst[i + 3] = (__bf16)f.w;
}

// ------------------- We [E][k][n] -> WeT bf16 [E][n][k] ---------------------
__global__ void k_wet(const float* __restrict__ We, __bf16* __restrict__ wet) {
  __shared__ float tile[32][33];
  const int e = blockIdx.z;
  const int kb = blockIdx.x * 32, nb = blockIdx.y * 32;
  const int tx = threadIdx.x, ty = threadIdx.y;  // 32 x 8
  const float* src = We + (size_t)e * Hh * Hh;
#pragma unroll
  for (int r = 0; r < 32; r += 8)
    tile[ty + r][tx] = src[(size_t)(kb + ty + r) * Hh + nb + tx];
  __syncthreads();
  __bf16* dst = wet + (size_t)e * Hh * Hh;
#pragma unroll
  for (int r = 0; r < 32; r += 8)
    dst[(size_t)(nb + ty + r) * Hh + kb + tx] = (__bf16)tile[tx][ty + r];
}

// --------------------------- mean_x[h] -------------------------------------
__global__ __launch_bounds__(256) void k_meanx(const float* __restrict__ x,
                                               float* __restrict__ mx) {
  const int h = blockIdx.x * 256 + threadIdx.x;
  float s = 0.f;
  for (int m = 0; m < Mm; ++m) s += x[(size_t)m * Hh + h];
  mx[h] = s * (1.f / Mm);
}

// ---- router: softmax gate, top-2 renorm, deterministic LB-loss partials ----
__global__ __launch_bounds__(256) void k_router(const float* __restrict__ x,
                                                const float* __restrict__ Wg,
                                                const float* __restrict__ bg,
                                                float* __restrict__ rout,
                                                float* __restrict__ part) {
  __shared__ float red[256];
  const int tid = threadIdx.x;
  const int t = blockIdx.x * 256 + tid;
  float l0 = bg[0], l1 = bg[1], l2 = bg[2], l3 = bg[3];
  const float* xr = x + (size_t)t * Hh;
  for (int h = 0; h < Hh; ++h) {
    const float xv = xr[h];
    l0 += xv * Wg[h * 4 + 0];
    l1 += xv * Wg[h * 4 + 1];
    l2 += xv * Wg[h * 4 + 2];
    l3 += xv * Wg[h * 4 + 3];
  }
  float l[4] = {l0, l1, l2, l3};
  const float mx = fmaxf(fmaxf(l0, l1), fmaxf(l2, l3));
  float p[4];
  float sum = 0.f;
#pragma unroll
  for (int e = 0; e < 4; ++e) { p[e] = __expf(l[e] - mx); sum += p[e]; }
  const float inv = 1.f / sum;
#pragma unroll
  for (int e = 0; e < 4; ++e) p[e] *= inv;
  // top-2, ties -> lower index (matches jax.lax.top_k)
  int i0 = 0;
  for (int e = 1; e < 4; ++e) if (p[e] > p[i0]) i0 = e;
  int i1 = (i0 == 0) ? 1 : 0;
  for (int e = 0; e < 4; ++e) if (e != i0 && p[e] > p[i1]) i1 = e;
  const float rsum = p[i0] + p[i1];
  float w[4] = {0.f, 0.f, 0.f, 0.f};
  w[i0] = p[i0] / rsum;
  w[i1] = p[i1] / rsum;
#pragma unroll
  for (int e = 0; e < 4; ++e) rout[t * 4 + e] = w[e];
  // deterministic block reductions: 4 prob sums + 4 top-2 selection counts
  for (int q = 0; q < 8; ++q) {
    const float val = (q < 4) ? p[q]
                              : ((q - 4 == i0 ? 1.f : 0.f) + (q - 4 == i1 ? 1.f : 0.f));
    __syncthreads();
    red[tid] = val;
    __syncthreads();
    for (int s = 128; s > 0; s >>= 1) {
      if (tid < s) red[tid] += red[tid + s];
      __syncthreads();
    }
    if (tid == 0) part[blockIdx.x * 8 + q] = red[0];
  }
}

// ------------- xae[t][e*4+r] = 0.25 * sum_h x[t,h]*Ae[e,h,r] ----------------
__global__ __launch_bounds__(256) void k_xae(const float* __restrict__ x,
                                             const float* __restrict__ Ae,
                                             float* __restrict__ xae) {
  const int gid = blockIdx.x * 256 + threadIdx.x;
  const int t = gid >> 4, er = gid & 15;
  const float* xr = x + (size_t)t * Hh;
  const float* A = Ae + (size_t)(er >> 2) * Hh * Rr + (er & 3);
  float acc = 0.f;
  for (int h = 0; h < Hh; ++h) acc += xr[h] * A[h * 4];
  xae[gid] = acc * 0.25f;  // fold ALPHA/rank
}

// ----- per-expert: exact top-512 membership + mean + folded value-LoRA ------
__global__ __launch_bounds__(256) void k_topk(const float* __restrict__ imp,
                                              const float* __restrict__ cv,
                                              const float* __restrict__ Av,
                                              const float* __restrict__ Bv,
                                              float* __restrict__ meanvec) {
  __shared__ float simp[Cc];
  __shared__ int ssel[Cc];
  __shared__ float smean[Hh];
  __shared__ float smav[4];
  const int e = blockIdx.x, tid = threadIdx.x;
#pragma unroll
  for (int j = 0; j < 4; ++j) simp[tid + 256 * j] = imp[e * Cc + tid + 256 * j];
  __syncthreads();
#pragma unroll
  for (int j = 0; j < 4; ++j) {
    const int c = tid + 256 * j;
    const float vi = simp[c];
    int rank = 0;
    for (int cc = 0; cc < Cc; ++cc) {
      const float vc = simp[cc];
      rank += (vc > vi) || (vc == vi && cc < c);  // top_k tie order
    }
    ssel[c] = (rank < KEEPc) ? 1 : 0;
  }
  __syncthreads();
  float a0 = 0.f, a1 = 0.f, a2 = 0.f, a3 = 0.f;
  const float* base = cv + (size_t)e * Cc * Hh;
  for (int c = 0; c < Cc; ++c) {
    if (ssel[c]) {
      const float* r = base + (size_t)c * Hh + tid;
      a0 += r[0];
      a1 += r[256];
      a2 += r[512];
      a3 += r[768];
    }
  }
  const float s = 1.f / KEEPc;
  smean[tid] = a0 * s;
  smean[tid + 256] = a1 * s;
  smean[tid + 512] = a2 * s;
  smean[tid + 768] = a3 * s;
  __syncthreads();
  if (tid < 4) {
    float a = 0.f;
    const float* Ap = Av + (size_t)e * Hh * Rr + tid;
    for (int h = 0; h < Hh; ++h) a += smean[h] * Ap[h * 4];
    smav[tid] = a;
  }
  __syncthreads();
  const float* Bp = Bv + (size_t)e * Rr * Hh;
#pragma unroll
  for (int j = 0; j < 4; ++j) {
    const int h = tid + 256 * j;
    meanvec[e * Hh + h] =
        smean[h] + 0.25f * (smav[0] * Bp[h] + smav[1] * Bp[Hh + h] +
                            smav[2] * Bp[2 * Hh + h] + smav[3] * Bp[3 * Hh + h]);
  }
}

// ------------------- fused MoE GEMM (the hot kernel) ------------------------
// grid (Mm/128, Hh/64), 256 thr = 8 waves. Wave w: rows [w*16, w*16+16),
// 4 N-subtiles of 16. Flat 128-step (expert, k-block) loop; B tile (64x32
// bf16, 80B LDS row stride: 16B-aligned, conflict-free b128) double-buffered
// and staged one step ahead (async-to-LDS when available); A fragment
// pipelined one step ahead in registers. One barrier per K-block.
__global__ __launch_bounds__(256) void k_moe_gemm(
    const __bf16* __restrict__ xb,    // [Mm][Hh] bf16
    const __bf16* __restrict__ wet,   // [Ee][n][k] bf16 (We transposed)
    const float* __restrict__ rout,   // [Mm][Ee]
    const float* __restrict__ xae,    // [Mm][16] (pre-scaled by alpha/rank)
    const float* __restrict__ be,     // [Ee][Hh]
    const float* __restrict__ Bel,    // Be LoRA [Ee][Rr][Hh]
    const float* __restrict__ meanvec,// [Ee][Hh]
    float* __restrict__ out) {        // [Mm][Hh]
  __shared__ __bf16 sB[2 * 64 * 40];  // two 64x32 tiles, row stride 40
  __shared__ float sW[128 * 4];
  __shared__ float sC[128 * 16];

  const int tid = threadIdx.x;
  const int wave = tid >> 5;
  const int lane = tid & 31;
  const int rlo = lane & 15;
  const int kh = lane >> 4;
  const int mbase = blockIdx.x * 128;
  const int nbase = blockIdx.y * 64;

  for (int idx = tid; idx < 512; idx += 256) sW[idx] = rout[mbase * 4 + idx];
  for (int idx = tid; idx < 2048; idx += 256) {
    const int t = idx >> 4, er = idx & 15;
    sC[idx] = rout[(mbase + t) * 4 + (er >> 2)] * xae[(mbase + t) * 16 + er];
  }

  v8f facc[4], acc[4];
#pragma unroll
  for (int j = 0; j < 4; ++j) { facc[j] = vzero8(); acc[j] = vzero8(); }

  const int arow = mbase + wave * 16 + rlo;
  const __bf16* aptr = xb + (size_t)arow * Hh;
  const int brow = tid >> 2;        // 0..63 : n within tile
  const int bchnk = (tid & 3) * 8;  // k chunk within 32
  // per-thread staging source for flat step t: wet[e(t)][nbase+brow][kb(t)*32+bchnk]
  const __bf16* wsrc = wet + ((size_t)(nbase + brow)) * Hh + bchnk;

  // ---- prologue: stage tile for t=0 into buffer 0 ----
#if USE_ASYNC_LDS
  __builtin_amdgcn_global_load_async_to_lds_b128(
      to_glob(wsrc), to_lds(&sB[brow * 40 + bchnk]), 0, 0);
  __builtin_amdgcn_s_wait_asynccnt(0);
#else
  *(v8bf*)(&sB[brow * 40 + bchnk]) = *(const v8bf*)wsrc;
#endif
  __syncthreads();
  v16bf af = ld16x2(aptr + kh * 8, aptr + kh * 8 + 16);  // A for t=0

  for (int t = 0; t < 128; ++t) {
    const int cur = (t & 1) * 2560;
    const int nxt = ((t + 1) & 1) * 2560;
    const int tn = (t + 1) & 127;  // wraps harmlessly at the end
    const __bf16* srcn =
        wsrc + (size_t)(tn >> 5) * Hh * Hh + (size_t)(tn & 31) * 32;

    // ---- stage B tile for step t+1 into the other buffer ----
#if USE_ASYNC_LDS
    __builtin_amdgcn_global_load_async_to_lds_b128(
        to_glob(srcn), to_lds(&sB[nxt + brow * 40 + bchnk]), 0, 0);
#else
    const v8bf breg = *(const v8bf*)srcn;
#endif
    // ---- A fragment for step t+1 (registers) ----
    const int k0n = (tn & 31) * 32 + kh * 8;
    const v16bf afn = ld16x2(aptr + k0n, aptr + k0n + 16);
    __builtin_prefetch(aptr + (((t + 2) & 31) * 32) + kh * 8, 0, 1);

    // ---- compute step t from buffer cur ----
#pragma unroll
    for (int j = 0; j < 4; ++j) {
      // B fragment: lane<16 -> cols n, K 0..7,16..23; lane>=16 -> K 8..15,24..31
      const __bf16* bp = &sB[cur + (j * 16 + rlo) * 40 + kh * 8];
      const v16bf bf = ld16x2(bp, bp + 16);
      acc[j] = __builtin_amdgcn_wmma_f32_16x16x32_bf16(
          false, af, false, bf, (short)0, acc[j], false, false);
    }

    // ---- expert boundary: apply routing weight, reset accumulators ----
    if ((t & 31) == 31) {
      const int e = t >> 5;
      float wgt[8];
#pragma unroll
      for (int i = 0; i < 8; ++i)
        wgt[i] = sW[(wave * 16 + kh * 8 + i) * 4 + e];
#pragma unroll
      for (int j = 0; j < 4; ++j)
#pragma unroll
        for (int i = 0; i < 8; ++i) {
          facc[j][i] += wgt[i] * acc[j][i];
          acc[j][i] = 0.f;
        }
    }

    // ---- retire staging for t+1, release buffer cur ----
#if USE_ASYNC_LDS
    __builtin_amdgcn_s_wait_asynccnt(0);
#else
    *(v8bf*)(&sB[nxt + brow * 40 + bchnk]) = breg;
#endif
    __syncthreads();
    af = afn;
  }

  // epilogue: bias + cached-value mean + rank-4 LoRA, routing-weighted
#pragma unroll
  for (int j = 0; j < 4; ++j) {
    const int h = nbase + j * 16 + rlo;
    float bmv[4], beh[16];
#pragma unroll
    for (int e = 0; e < 4; ++e) bmv[e] = be[e * Hh + h] + meanvec[e * Hh + h];
#pragma unroll
    for (int er = 0; er < 16; ++er)
      beh[er] = Bel[(size_t)((er >> 2) * Rr + (er & 3)) * Hh + h];
#pragma unroll
    for (int i = 0; i < 8; ++i) {
      const int ml = wave * 16 + kh * 8 + i;
      float v = facc[j][i];
#pragma unroll
      for (int e = 0; e < 4; ++e) v += sW[ml * 4 + e] * bmv[e];
      const float* cf = &sC[ml * 16];
#pragma unroll
      for (int er = 0; er < 16; ++er) v += cf[er] * beh[er];
      out[(size_t)(mbase + ml) * Hh + h] = v;
    }
  }
}

// -------- new_values row0: mean_x@We + be + 0.25*(mean_x@Ae)@Be + meanvec ---
__global__ __launch_bounds__(256) void k_nvrow(const float* __restrict__ mx,
                                               const float* __restrict__ We,
                                               const float* __restrict__ be,
                                               const float* __restrict__ Ae,
                                               const float* __restrict__ Bel,
                                               const float* __restrict__ meanvec,
                                               float* __restrict__ nv) {
  const int gid = blockIdx.x * 256 + threadIdx.x;  // 0..4095
  const int e = gid >> 10, h = gid & 1023;
  const float* W = We + (size_t)e * Hh * Hh + h;
  const float* A = Ae + (size_t)e * Hh * Rr;
  float dot = 0.f, ma0 = 0.f, ma1 = 0.f, ma2 = 0.f, ma3 = 0.f;
  for (int k = 0; k < Hh; ++k) {
    const float m = mx[k];
    dot += m * W[(size_t)k * Hh];
    ma0 += m * A[k * 4 + 0];
    ma1 += m * A[k * 4 + 1];
    ma2 += m * A[k * 4 + 2];
    ma3 += m * A[k * 4 + 3];
  }
  const float* Bp = Bel + (size_t)e * Rr * Hh + h;
  const float lr = ma0 * Bp[0] + ma1 * Bp[Hh] + ma2 * Bp[2 * Hh] + ma3 * Bp[3 * Hh];
  nv[(size_t)e * Cc * Hh + h] = dot + be[e * Hh + h] + 0.25f * lr + meanvec[e * Hh + h];
}

// ---------------- lb_loss, new_keys row0, new_imp row0 ----------------------
__global__ __launch_bounds__(256) void k_fin(const float* __restrict__ part,
                                             const float* __restrict__ mx,
                                             float* __restrict__ lb,
                                             float* __restrict__ nk,
                                             float* __restrict__ ni) {
  const int h = blockIdx.x * 256 + threadIdx.x;
  const float v = mx[h];
#pragma unroll
  for (int e = 0; e < Ee; ++e) nk[(size_t)e * Cc * Hh + h] = v;
  if (h == 0) {
    float sp[4] = {0.f, 0.f, 0.f, 0.f}, sc[4] = {0.f, 0.f, 0.f, 0.f};
    for (int b = 0; b < Mm / 256; ++b)
      for (int e = 0; e < 4; ++e) {
        sp[e] += part[b * 8 + e];
        sc[e] += part[b * 8 + 4 + e];
      }
    float acc = 0.f;
    for (int e = 0; e < 4; ++e) acc += (sp[e] / Mm) * (sc[e] / (Mm * 2.f));
    lb[0] = 4.f * acc;
    // importance.mean() == 1.0 exactly: softmax rows sum to 1.
    for (int e = 0; e < 4; ++e) ni[e * Cc] = 1.0f;
  }
}

// ---------------------------------------------------------------------------
extern "C" void kernel_launch(void* const* d_in, const int* in_sizes, int n_in,
                              void* d_out, int out_size, void* d_ws, size_t ws_size,
                              hipStream_t stream) {
  const float* x  = (const float*)d_in[0];
  // d_in[1] (query), d_in[4..11] (Wq,bq,Wk,bk,Aq,Bq,Ak,Bk) provably unused.
  const float* Wg = (const float*)d_in[2];
  const float* bg = (const float*)d_in[3];
  const float* We = (const float*)d_in[12];
  const float* be = (const float*)d_in[13];
  const float* Ae = (const float*)d_in[14];
  const float* Be = (const float*)d_in[15];
  const float* Av = (const float*)d_in[16];
  const float* Bv = (const float*)d_in[17];
  const float* ck = (const float*)d_in[18];
  const float* cv = (const float*)d_in[19];
  const float* ci = (const float*)d_in[20];

  float* out = (float*)d_out;
  float* eo = out;                              // [Mm][Hh]
  float* lb = out + (size_t)Mm * Hh;            // [1]
  float* nk = lb + 1;                           // [Ee][Cc][Hh]
  float* nv = nk + (size_t)Ee * Cc * Hh;        // [Ee][Cc][Hh]
  float* ni = nv + (size_t)Ee * Cc * Hh;        // [Ee][Cc]

  // workspace layout (~41.3 MiB, all regions fully rewritten every call)
  char* w = (char*)d_ws;
  __bf16* xb   = (__bf16*)(w + 0);          // 33,554,432 B
  __bf16* wet  = (__bf16*)(w + 33554432);   //  8,388,608 B
  float* rout  = (float*)(w + 41943040);    //    262,144 B
  float* xae   = (float*)(w + 42205184);    //  1,048,576 B
  float* mx    = (float*)(w + 43253760);    //      4,096 B
  float* mvec  = (float*)(w + 43257856);    //     16,384 B
  float* part  = (float*)(w + 43274240);    //      2,048 B

  k_cvt_bf16<<<(Mm * Hh) / 1024, 256, 0, stream>>>(x, xb);
  k_wet<<<dim3(Hh / 32, Hh / 32, Ee), dim3(32, 8), 0, stream>>>(We, wet);
  k_meanx<<<Hh / 256, 256, 0, stream>>>(x, mx);
  k_router<<<Mm / 256, 256, 0, stream>>>(x, Wg, bg, rout, part);
  k_xae<<<(Mm * 16) / 256, 256, 0, stream>>>(x, Ae, xae);
  k_topk<<<Ee, 256, 0, stream>>>(ci, cv, Av, Bv, mvec);
  k_moe_gemm<<<dim3(Mm / 128, Hh / 64), 256, 0, stream>>>(xb, wet, rout, xae,
                                                          be, Be, mvec, eo);
  (void)hipMemcpyAsync(nk, ck, sizeof(float) * (size_t)Ee * Cc * Hh,
                       hipMemcpyDeviceToDevice, stream);
  (void)hipMemcpyAsync(nv, cv, sizeof(float) * (size_t)Ee * Cc * Hh,
                       hipMemcpyDeviceToDevice, stream);
  (void)hipMemcpyAsync(ni, ci, sizeof(float) * (size_t)Ee * Cc,
                       hipMemcpyDeviceToDevice, stream);
  k_nvrow<<<(Ee * Hh) / 256, 256, 0, stream>>>(mx, We, be, Ae, Be, mvec, nv);
  k_fin<<<Hh / 256, 256, 0, stream>>>(part, mx, lb, nk, ni);

  (void)in_sizes; (void)n_in; (void)out_size; (void)ws_size;
}
